// EmaAggregator_66692252172491
// MI455X (gfx1250) — matrix operand order repaired and checked by reading, not compiled, not run on hardware
//
#include <hip/hip_runtime.h>

// ---------------------------------------------------------------------------
// MI455X / gfx1250 implementation.
// Heavy GEMMs via v_wmma_f32_16x16x32_bf16 (bf16 operands, f32 accum).
// LayerNorm over the 4096-wide hidden is folded into the second GEMM as a
// per-row affine so 'h' is staged once (bf16) instead of twice.
// All weight operands are pre-transposed (N x K) so both GEMM tiles are
// identical strided rectangles -> staged by the Tensor Data Mover
// (tensor_load_to_lds), double-buffered, with s_wait_tensorcnt pipelining.
// ---------------------------------------------------------------------------

typedef unsigned short u16;
typedef unsigned int   u32;
typedef unsigned long long u64;
typedef __attribute__((ext_vector_type(16))) __bf16 v16bf;
typedef __attribute__((ext_vector_type(8)))  float  v8f;
typedef __attribute__((ext_vector_type(4)))  unsigned int v4u;
typedef __attribute__((ext_vector_type(8)))  int v8i;
typedef __attribute__((ext_vector_type(4)))  int v4i;

#define BM 128
#define BN 128
#define BK 32
#define TPB 256

#if __has_builtin(__builtin_amdgcn_global_load_async_to_lds_b128) && \
    __has_builtin(__builtin_amdgcn_s_wait_asynccnt)
#define USE_ASYNC_LDS 1
#else
#define USE_ASYNC_LDS 0
#endif

#if __has_builtin(__builtin_amdgcn_tensor_load_to_lds) && \
    __has_builtin(__builtin_amdgcn_s_wait_tensorcnt)
#define USE_TDM 1
#else
#define USE_TDM 0
#endif

// Async copy operand types per compiler diagnostic: global/local v4i pointers.
typedef __attribute__((address_space(1))) v4i* gv4i_p;
typedef __attribute__((address_space(3))) v4i* lv4i_p;
#define AS3V(p)  ((__attribute__((address_space(3))) void*)(void*)(p))

static __device__ __forceinline__ u16 f32_to_bf16(float f) {
  union { float f; u32 u; } x; x.f = f;
  u32 u = x.u;
  u += 0x7fffu + ((u >> 16) & 1u);   // round-to-nearest-even
  return (u16)(u >> 16);
}

static __device__ __forceinline__ float gelu_exact(float x) {
  return 0.5f * x * (1.0f + erff(x * 0.70710678118654752f));
}

#if USE_TDM
// One 2D TDM transfer: tile_k x tile_rows bf16 tile, row stride strideElems,
// from global byte address gaddr into LDS offset ldsOff.
// Descriptor per CDNA5 ISA (data_size=2B).
static __device__ __forceinline__ void tdm_load_2d(const void* gaddr, u32 ldsOff,
                                                   u32 tileK, u32 tileRows,
                                                   u32 strideElems) {
  u64 ga = (u64)(uintptr_t)gaddr;
  v4u g0;
  g0.x = 1u;                                            // count=1, user D#
  g0.y = ldsOff;                                        // lds_addr
  g0.z = (u32)ga;                                       // global_addr[31:0]
  g0.w = ((u32)(ga >> 32) & 0x01FFFFFFu) | (2u << 30);  // addr[56:32] | type=2
  v8i g1;
  g1[0] = (int)(1u << 16);                              // wg_mask=0, 2B elems
  g1[1] = (int)((tileK & 0xFFFFu) << 16);               // tensor_dim0 lo
  g1[2] = (int)(((tileK >> 16) & 0xFFFFu) | ((tileRows & 0xFFFFu) << 16));
  g1[3] = (int)(((tileRows >> 16) & 0xFFFFu) | ((tileK & 0xFFFFu) << 16));
  g1[4] = (int)(tileRows & 0xFFFFu);                    // tile_dim1, dim2=0
  g1[5] = (int)strideElems;                             // tensor_dim0_stride
  g1[6] = 0;
  g1[7] = 0;
  v4i z4 = {0, 0, 0, 0};
#if __clang_major__ >= 23
  v8i z8 = {0, 0, 0, 0, 0, 0, 0, 0};
  __builtin_amdgcn_tensor_load_to_lds(g0, g1, z4, z4, z8, 0);
#else
  __builtin_amdgcn_tensor_load_to_lds(g0, g1, z4, z4, 0);
#endif
}
#endif

// ----------------------------- elementwise prep ----------------------------

__global__ void k_f32_to_bf16(const float* __restrict__ src, u16* __restrict__ dst,
                              long long n) {
  long long i = (long long)blockIdx.x * blockDim.x + threadIdx.x;
  if (i < n) dst[i] = f32_to_bf16(src[i]);
}

// Convert f32 [R x C] -> bf16 transposed [C x R], optional per-row scale.
// Dims are multiples of 32 here.
__global__ __launch_bounds__(256)
void k_convT(const float* __restrict__ src, const float* __restrict__ rowScale,
             u16* __restrict__ dst, int R, int C) {
  __shared__ u16 tile[32][33];
  int bc = blockIdx.x * 32;        // source column block
  int br = blockIdx.y * 32;        // source row block
  int tx = threadIdx.x & 31;
  int ty = threadIdx.x >> 5;       // 0..7
  #pragma unroll
  for (int j = 0; j < 32; j += 8) {
    int r = br + ty + j, c = bc + tx;
    float v = src[(size_t)r * C + c];
    if (rowScale) v *= rowScale[r];
    tile[ty + j][tx] = f32_to_bf16(v);
  }
  __syncthreads();
  #pragma unroll
  for (int j = 0; j < 32; j += 8) {
    int rr = bc + ty + j;          // dst row = src col
    int cc = br + tx;              // dst col = src row
    dst[(size_t)rr * R + cc] = tile[tx][ty + j];
  }
}

__global__ void k_c0c1(const float* __restrict__ W2, const float* __restrict__ g1,
                       const float* __restrict__ be1, float* __restrict__ c0,
                       float* __restrict__ c1, int F, int D) {
  int d = blockIdx.x * blockDim.x + threadIdx.x;
  if (d >= D) return;
  float a = 0.f, b = 0.f;
  for (int f = 0; f < F; ++f) {
    float w = W2[(long long)f * D + d];
    a += be1[f] * w;
    b += g1[f] * w;
  }
  c0[d] = a; c1[d] = b;
}

// ------------------------------- WMMA GEMM ---------------------------------
// C[M,N] = A[M,K] * Bt[N,K]^T (both bf16 row-major; Bt is pre-transposed B).
// mode 0: v = gelu(acc + bias[n]); store bf16 to outBf; accumulate row stats.
// mode 1: v = rstd[m]*acc + bias[n] + c0[n] - mean[m]*rstd[m]*c1[n];
//         store f32 to outF; accumulate row stats.   (LN1 folded GEMM2)
// mode 2: v = gelu(acc + bias[n]); store f32 to outF; accumulate row stats.

union FragBf { v16bf v; uint4 q[2]; };

// Fragment loads + 8 WMMAs for one BK=32 step.
// A 16-bit 16x32 layout: lanes 0-15 hold K{0..7,16..23}, 16-31 K{8..15,24..31}.
// B 32x16: lanes 0-15 hold K{0..7,16..23}, 16-31 K{8..15,24..31} (col = lane&15).
static __device__ __forceinline__ void wmma_step(const u16* __restrict__ pa,
                                                 const u16* __restrict__ pb,
                                                 int wm, int wn, int half, int l15,
                                                 v8f acc[2][4]) {
  FragBf af[2], bfr[4];
  #pragma unroll
  for (int mi = 0; mi < 2; ++mi) {
    const u16* p = &pa[(wm + mi * 16 + l15) * BK + half * 8];
    af[mi].q[0] = *(const uint4*)p;
    af[mi].q[1] = *(const uint4*)(p + 16);
  }
  #pragma unroll
  for (int ni = 0; ni < 4; ++ni) {
    const u16* p = &pb[(wn + ni * 16 + l15) * BK + half * 8];
    bfr[ni].q[0] = *(const uint4*)p;
    bfr[ni].q[1] = *(const uint4*)(p + 16);
  }
  #pragma unroll
  for (int mi = 0; mi < 2; ++mi)
    #pragma unroll
    for (int ni = 0; ni < 4; ++ni)
      acc[mi][ni] = __builtin_amdgcn_wmma_f32_16x16x32_bf16(
          false, af[mi].v, false, bfr[ni].v, (short)0, acc[mi][ni],
          false, false);
}

__global__ __launch_bounds__(TPB)
void k_gemm_bf16(const u16* __restrict__ A, const u16* __restrict__ Bt,
                 float* __restrict__ outF, u16* __restrict__ outBf,
                 const float* __restrict__ bias,
                 const float* __restrict__ c0, const float* __restrict__ c1,
                 const float* __restrict__ rowMean, const float* __restrict__ rowRstd,
                 float* __restrict__ rowSum, float* __restrict__ rowSumSq,
                 int M, int N, int K, int mode) {
  __shared__ __align__(16) u16 sA[2][BM * BK];   // [m][k] row-major, 2 buffers
  __shared__ __align__(16) u16 sB[2][BN * BK];   // [n][k] row-major, 2 buffers

  const int t    = threadIdx.x;
  const int bm   = blockIdx.y * BM;
  const int bn   = blockIdx.x * BN;
  const int lane = t & 31;
  const int wave = t >> 5;
  const int wm   = (wave >> 1) * 32;   // 4x2 wave grid, each wave 32x64
  const int wn   = (wave & 1) * 64;
  const int half = lane >> 4;
  const int l15  = lane & 15;
  const int nk   = K / BK;

  v8f zf;
  #pragma unroll
  for (int i = 0; i < 8; ++i) zf[i] = 0.f;
  v8f acc[2][4];
  #pragma unroll
  for (int mi = 0; mi < 2; ++mi)
    #pragma unroll
    for (int ni = 0; ni < 4; ++ni) acc[mi][ni] = zf;

  const u16* Abase = A  + (size_t)bm * K;
  const u16* Bbase = Bt + (size_t)bn * K;

#if USE_TDM
  // ---- TDM double-buffered pipeline: one DMA per tile per k-step ----
  const u32 ldsA0 = (u32)(uintptr_t)AS3V(&sA[0][0]);
  const u32 ldsA1 = (u32)(uintptr_t)AS3V(&sA[1][0]);
  const u32 ldsB0 = (u32)(uintptr_t)AS3V(&sB[0][0]);
  const u32 ldsB1 = (u32)(uintptr_t)AS3V(&sB[1][0]);
  if (wave == 0) {
    tdm_load_2d(Abase, ldsA0, BK, BM, K);
    tdm_load_2d(Bbase, ldsB0, BK, BN, K);
    if (nk > 1) {
      tdm_load_2d(Abase + BK, ldsA1, BK, BM, K);
      tdm_load_2d(Bbase + BK, ldsB1, BK, BN, K);
    }
  }
  for (int i = 0; i < nk; ++i) {
    const int buf = i & 1;
    if (wave == 0) {
      if (i + 1 < nk) __builtin_amdgcn_s_wait_tensorcnt(2);  // pair i done
      else            __builtin_amdgcn_s_wait_tensorcnt(0);
    }
    __syncthreads();                       // publish buf to all waves
    wmma_step(sA[buf], sB[buf], wm, wn, half, l15, acc);
    __syncthreads();                       // all waves done reading buf
    if (wave == 0 && i + 2 < nk) {
      size_t k2 = (size_t)(i + 2) * BK;
      tdm_load_2d(Abase + k2, buf ? ldsA1 : ldsA0, BK, BM, K);
      tdm_load_2d(Bbase + k2, buf ? ldsB1 : ldsB0, BK, BN, K);
      if (i + 4 < nk && lane == 0) {       // stream HBM ahead of the TDM
        __builtin_prefetch(Abase + (size_t)(i + 4) * BK, 0, 1);
        __builtin_prefetch(Bbase + (size_t)(i + 4) * BK, 0, 1);
      }
    }
  }
#else
  // ---- fallback: async b128 copies, single buffer ----
  const int am = t >> 1;           // tile row, 16B chunk index aq (+2 per iter)
  const int aq = t & 1;
  for (int i = 0; i < nk; ++i) {
    int k0 = i * BK;
    #pragma unroll
    for (int it = 0; it < 2; ++it) {
      int q = aq + 2 * it;
      const u16* ga = Abase + (size_t)am * K + k0 + q * 8;
      const u16* gb = Bbase + (size_t)am * K + k0 + q * 8;
#if USE_ASYNC_LDS
      __builtin_amdgcn_global_load_async_to_lds_b128(
          (gv4i_p)(v4i*)const_cast<u16*>(ga),
          (lv4i_p)(v4i*)&sA[0][am * BK + q * 8], 0, 0);
      __builtin_amdgcn_global_load_async_to_lds_b128(
          (gv4i_p)(v4i*)const_cast<u16*>(gb),
          (lv4i_p)(v4i*)&sB[0][am * BK + q * 8], 0, 0);
#else
      *(uint4*)&sA[0][am * BK + q * 8] = *(const uint4*)ga;
      *(uint4*)&sB[0][am * BK + q * 8] = *(const uint4*)gb;
#endif
    }
    if (k0 + BK < K) {
      __builtin_prefetch(Abase + (size_t)am * K + k0 + BK, 0, 1);
      __builtin_prefetch(Bbase + (size_t)am * K + k0 + BK, 0, 1);
    }
#if USE_ASYNC_LDS
    __builtin_amdgcn_s_wait_asynccnt(0);
#endif
    __syncthreads();
    wmma_step(sA[0], sB[0], wm, wn, half, l15, acc);
    __syncthreads();
  }
#endif

  // Epilogue. C layout: VGPR r, lanes 0-15 -> M=r, lanes 16-31 -> M=r+8, N=lane&15.
  const bool md0 = (mode == 0), md1 = (mode == 1);
  #pragma unroll
  for (int mi = 0; mi < 2; ++mi) {
    #pragma unroll
    for (int r = 0; r < 8; ++r) {
      int m = bm + wm + mi * 16 + half * 8 + r;
      float mn = 0.f, rr = 0.f;
      if (md1) { mn = rowMean[m]; rr = rowRstd[m]; }
      float rs = 0.f, rq = 0.f;
      #pragma unroll
      for (int ni = 0; ni < 4; ++ni) {
        int n = bn + wn + ni * 16 + l15;
        float a = acc[mi][ni][r];
        float v;
        if (md1) v = rr * a + bias[n] + c0[n] - mn * rr * c1[n];
        else     v = gelu_exact(a + bias[n]);
        if (md0) outBf[(size_t)m * N + n] = f32_to_bf16(v);
        else     outF [(size_t)m * N + n] = v;
        rs += v; rq += v * v;
      }
      // Reduce the 16 lanes that share row m, then one f32 atomic per row.
      #pragma unroll
      for (int off = 1; off < 16; off <<= 1) {
        rs += __shfl_xor(rs, off, 32);
        rq += __shfl_xor(rq, off, 32);
      }
      if (l15 == 0) {
        atomicAdd(&rowSum[m], rs);
        atomicAdd(&rowSumSq[m], rq);
      }
    }
  }
}

// ------------------------------ LN plumbing --------------------------------

__global__ void k_finalize_stats(const float* __restrict__ sum,
                                 const float* __restrict__ sumsq,
                                 float* __restrict__ mean, float* __restrict__ rstd,
                                 int rows, float invN) {
  int r = blockIdx.x * blockDim.x + threadIdx.x;
  if (r >= rows) return;
  float m = sum[r] * invN;
  float v = sumsq[r] * invN - m * m;
  mean[r] = m;
  rstd[r] = rsqrtf(v + 1e-5f);
}

__global__ void k_ln_apply(float* __restrict__ buf, const float* __restrict__ mean,
                           const float* __restrict__ rstd, const float* __restrict__ g,
                           const float* __restrict__ b, long long total, int D) {
  long long i = (long long)blockIdx.x * blockDim.x + threadIdx.x;
  if (i >= total) return;
  long long row = i / D;
  int d = (int)(i - row * D);
  buf[i] = (buf[i] - mean[row]) * rstd[row] * g[d] + b[d];
}

// ----------------------------- EMA recurrence ------------------------------

__global__ void k_ema_group(const float* __restrict__ phi,
                            const float* __restrict__ alpha4,
                            const float* __restrict__ sIn,
                            float* __restrict__ outBase,
                            u16* __restrict__ comb, int B, int D) {
  int e = blockIdx.x * blockDim.x + threadIdx.x;
  long long BD = (long long)B * D;
  if (e >= BD) return;
  int b = e / D;
  int d = e - b * D;
  float s = sIn[e];
  float ck = s;
  #pragma unroll
  for (int l = 0; l < 4; ++l) {
    float a = 1.0f / (1.0f + __expf(-alpha4[l]));
    s = (1.0f - a) * s + a * phi[(long long)l * BD + e];
    outBase[(long long)l * BD + e] = s;
    if (l == 1) ck = s;   // checkpoint = s_list[-2] at the tree step
  }
  comb[(size_t)b * (2 * D) + d]     = f32_to_bf16(s);
  comb[(size_t)b * (2 * D) + D + d] = f32_to_bf16(ck);
}

__global__ void k_tree_finalize(const float* __restrict__ pre,
                                const float* __restrict__ mean,
                                const float* __restrict__ rstd,
                                const float* __restrict__ g, const float* __restrict__ bb,
                                float* __restrict__ sbuf, float* __restrict__ outLayer,
                                int B, int D) {
  int e = blockIdx.x * blockDim.x + threadIdx.x;
  if (e >= B * D) return;
  int b = e / D;
  int d = e - b * D;
  float v = (pre[e] - mean[b]) * rstd[b] * g[d] + bb[d];
  sbuf[e] = v;
  outLayer[e] = v;
}

// ------------------------------- launcher ----------------------------------

extern "C" void kernel_launch(void* const* d_in, const int* in_sizes, int n_in,
                              void* d_out, int out_size, void* d_ws, size_t ws_size,
                              hipStream_t stream) {
  (void)in_sizes; (void)n_in; (void)out_size; (void)ws_size;
  const int D = 1024, F = 4096, L = 32, B = 1024;
  const int M = L * B;           // 32768 rows through the phi MLP
  const int TC = 4;              // tree_checkpoint (reference setup value)

  const float* z     = (const float*)d_in[0];
  const float* sInit = (const float*)d_in[1];
  const float* alpha = (const float*)d_in[2];
  const float* W1    = (const float*)d_in[3];
  const float* b1    = (const float*)d_in[4];
  const float* g1    = (const float*)d_in[5];
  const float* be1   = (const float*)d_in[6];
  const float* W2    = (const float*)d_in[7];
  const float* b2    = (const float*)d_in[8];
  const float* g2    = (const float*)d_in[9];
  const float* be2   = (const float*)d_in[10];
  const float* Wt    = (const float*)d_in[11];
  const float* bt    = (const float*)d_in[12];
  const float* gt    = (const float*)d_in[13];
  const float* bet   = (const float*)d_in[14];
  float* out = (float*)d_out;

  // Workspace carving (256B aligned).
  char* ws = (char*)d_ws;
  size_t off = 0;
  auto carve = [&](size_t bytes) -> char* {
    char* p = ws + off;
    off = (off + bytes + 255) & ~(size_t)255;
    return p;
  };
  u16*   zbf    = (u16*)  carve((size_t)M * D * 2);       //  67 MB
  u16*   w1t    = (u16*)  carve((size_t)F * D * 2);       // W1^T      [F x D]
  u16*   w2gt   = (u16*)  carve((size_t)D * F * 2);       // (g1.W2)^T [D x F]
  u16*   wtT    = (u16*)  carve((size_t)D * 2 * D * 2);   // Wt^T      [D x 2D]
  u16*   hbf    = (u16*)  carve((size_t)M * F * 2);       // 268 MB
  float* phi    = (float*)carve((size_t)M * D * 4);       // 134 MB
  float* stats1 = (float*)carve((size_t)2 * M * 4);       // sum | sumsq
  float* mean1  = (float*)carve((size_t)M * 4);
  float* rstd1  = (float*)carve((size_t)M * 4);
  float* stats2 = (float*)carve((size_t)2 * M * 4);
  float* mean2  = (float*)carve((size_t)M * 4);
  float* rstd2  = (float*)carve((size_t)M * 4);
  float* c0     = (float*)carve((size_t)D * 4);
  float* c1     = (float*)carve((size_t)D * 4);
  float* sbuf   = (float*)carve((size_t)B * D * 4);
  u16*   comb   = (u16*)  carve((size_t)B * 2 * D * 2);
  float* trpre  = (float*)carve((size_t)B * D * 4);
  float* stats3 = (float*)carve((size_t)2 * B * 4);
  float* mean3  = (float*)carve((size_t)B * 4);
  float* rstd3  = (float*)carve((size_t)B * 4);

  auto blk = [](long long n) { return (unsigned)((n + 255) / 256); };

  // 1. precision prep: z -> bf16; weights -> bf16 transposed (g1 folded into W2)
  k_f32_to_bf16<<<blk((long long)M * D), 256, 0, stream>>>(z, zbf, (long long)M * D);
  { dim3 g(F / 32, D / 32);     k_convT<<<g, 256, 0, stream>>>(W1, nullptr, w1t,  D,     F); }
  { dim3 g(D / 32, F / 32);     k_convT<<<g, 256, 0, stream>>>(W2, g1,      w2gt, F,     D); }
  { dim3 g(D / 32, 2 * D / 32); k_convT<<<g, 256, 0, stream>>>(Wt, nullptr, wtT,  2 * D, D); }
  k_c0c1<<<blk(D), 256, 0, stream>>>(W2, g1, be1, c0, c1, F, D);

  (void)hipMemsetAsync(stats1, 0, (size_t)2 * M * 4, stream);
  (void)hipMemsetAsync(stats2, 0, (size_t)2 * M * 4, stream);

  // 2. GEMM1: h = gelu(z@W1 + b1) -> bf16, + LN1 row stats
  {
    dim3 grid(F / BN, M / BM);
    k_gemm_bf16<<<grid, TPB, 0, stream>>>(zbf, w1t, nullptr, hbf, b1,
                                          nullptr, nullptr, nullptr, nullptr,
                                          stats1, stats1 + M, M, F, D, 0);
  }
  k_finalize_stats<<<blk(M), 256, 0, stream>>>(stats1, stats1 + M, mean1, rstd1,
                                               M, 1.0f / F);

  // 3. GEMM2 with LN1 folded: phi_pre = r*(h.g1)@W2 + b2 + c0 - m*r*c1, + LN2 stats
  {
    dim3 grid(D / BN, M / BM);
    k_gemm_bf16<<<grid, TPB, 0, stream>>>(hbf, w2gt, phi, nullptr, b2,
                                          c0, c1, mean1, rstd1,
                                          stats2, stats2 + M, M, D, F, 1);
  }
  k_finalize_stats<<<blk(M), 256, 0, stream>>>(stats2, stats2 + M, mean2, rstd2,
                                               M, 1.0f / D);
  k_ln_apply<<<blk((long long)M * D), 256, 0, stream>>>(phi, mean2, rstd2, g2, be2,
                                                        (long long)M * D, D);

  // 4. recurrence: 8 groups of 4 layers, tree MLP (WMMA) every 4th layer
  (void)hipMemcpyAsync(sbuf, sInit, (size_t)B * D * 4, hipMemcpyDeviceToDevice, stream);
  const long long BD = (long long)B * D;
  for (int g = 0; g < L / TC; ++g) {
    k_ema_group<<<blk(BD), 256, 0, stream>>>(phi + (long long)g * TC * BD,
                                             alpha + g * TC, sbuf,
                                             out + (long long)g * TC * BD,
                                             comb, B, D);
    (void)hipMemsetAsync(stats3, 0, (size_t)2 * B * 4, stream);
    {
      dim3 grid(D / BN, B / BM);
      k_gemm_bf16<<<grid, TPB, 0, stream>>>(comb, wtT, trpre, nullptr, bt,
                                            nullptr, nullptr, nullptr, nullptr,
                                            stats3, stats3 + B, B, D, 2 * D, 2);
    }
    k_finalize_stats<<<blk(B), 256, 0, stream>>>(stats3, stats3 + B, mean3, rstd3,
                                                 B, 1.0f / D);
    k_tree_finalize<<<blk(BD), 256, 0, stream>>>(trpre, mean3, rstd3, gt, bet, sbuf,
                                                 out + ((long long)g * TC + TC - 1) * BD,
                                                 B, D);
  }
}
